// EncodeParallel_74182675136897
// MI455X (gfx1250) — matrix operand order repaired and checked by reading, not compiled, not run on hardware
//
#include <hip/hip_runtime.h>
#include <hip/hip_bf16.h>

#define D_MODEL 1024
#define SEQ     2048
#define BATCH   4

typedef __attribute__((ext_vector_type(16))) __bf16 bhalf16;
typedef __attribute__((ext_vector_type(8)))  __bf16 bhalf8;
typedef __attribute__((ext_vector_type(8)))  float  v8f;

static __device__ inline v8f vzero8() {
  v8f z;
#pragma unroll
  for (int i = 0; i < 8; ++i) z[i] = 0.0f;
  return z;
}

// A-fragment (16x32, bf16): lane L holds row M=L&15; element j holds
// K = 16*(j>>3) + 8*h + (j&7), h = L>>4.  rowPtr = &row[kBase].
static __device__ inline bhalf16 load_afrag(const __bf16* rowPtr, int h) {
  bhalf8 lo = *reinterpret_cast<const bhalf8*>(rowPtr + 8 * h);
  bhalf8 hi = *reinterpret_cast<const bhalf8*>(rowPtr + 16 + 8 * h);
  bhalf16 r;
#pragma unroll
  for (int j = 0; j < 8; ++j) { r[j] = lo[j]; r[j + 8] = hi[j]; }
  return r;
}

// B-fragment (32x16, bf16): lane L holds column N=L&15; element j holds
// K = 16*(j>>3) + 8*h + (j&7), h = L>>4.  colPtr = &col_k_run[0] (contiguous in K).
static __device__ inline bhalf16 load_bfrag(const __bf16* colPtr, int h) {
  bhalf8 lo = *reinterpret_cast<const bhalf8*>(colPtr + 8 * h);
  bhalf8 hi = *reinterpret_cast<const bhalf8*>(colPtr + 16 + 8 * h);
  bhalf16 r;
#pragma unroll
  for (int j = 0; j < 8; ++j) { r[j] = lo[j]; r[j + 8] = hi[j]; }
  return r;
}

__global__ __launch_bounds__(256)
void cvt_f32_to_bf16(const float* __restrict__ src, unsigned short* __restrict__ dst, int n) {
  int i = blockIdx.x * blockDim.x + threadIdx.x;
  if (i < n) {
    __bf16 v = (__bf16)src[i];
    dst[i] = __builtin_bit_cast(unsigned short, v);
  }
}

// y = x @ W^T + b.  M = BATCH*SEQ, N = K = D_MODEL.  x,W in bf16 row-major.
// Wave tile 32x64 (8 f32 accum tiles); wave grid 4(M) x 2(N) => block tile 128x128.
// transposed==0: out[m][e] bf16 row-major.  transposed==1: out = Vt[b][e][s] bf16.
__global__ __launch_bounds__(256)
void qkv_gemm(const unsigned short* __restrict__ xw, const unsigned short* __restrict__ ww,
              const float* __restrict__ bias, unsigned short* __restrict__ outw, int transposed) {
  const __bf16* xb = reinterpret_cast<const __bf16*>(xw);
  const __bf16* W  = reinterpret_cast<const __bf16*>(ww);

  const int tid = threadIdx.x;
  const int lane = tid & 31, w = tid >> 5;
  const int l15 = lane & 15, h = lane >> 4;
  const int wm = w & 3, wn = w >> 2;                 // 4x2 wave grid
  const int m0 = blockIdx.x * 128 + wm * 32;         // 32 rows per wave
  const int e0 = blockIdx.y * 128 + wn * 64;         // 64 cols per wave

  v8f acc[2][4];
#pragma unroll
  for (int i = 0; i < 2; ++i)
#pragma unroll
    for (int j = 0; j < 4; ++j) acc[i][j] = vzero8();

  for (int k = 0; k < D_MODEL; k += 32) {
    bhalf16 a[2], bf[4];
#pragma unroll
    for (int i = 0; i < 2; ++i)
      a[i] = load_afrag(xb + (size_t)(m0 + 16 * i + l15) * D_MODEL + k, h);
#pragma unroll
    for (int j = 0; j < 4; ++j)
      bf[j] = load_bfrag(W + (size_t)(e0 + 16 * j + l15) * D_MODEL + k, h);
#pragma unroll
    for (int i = 0; i < 2; ++i)
#pragma unroll
      for (int j = 0; j < 4; ++j)
        acc[i][j] = __builtin_amdgcn_wmma_f32_16x16x32_bf16(
            false, a[i], false, bf[j], (short)0, acc[i][j], false, false);
  }

#pragma unroll
  for (int i = 0; i < 2; ++i)
#pragma unroll
    for (int j = 0; j < 4; ++j) {
      const int ecol = e0 + 16 * j + l15;
      const float bv = bias[ecol];
      if (!transposed) {
#pragma unroll
        for (int r = 0; r < 8; ++r) {
          const int m = m0 + 16 * i + r + 8 * h;
          __bf16 v = (__bf16)(acc[i][j][r] + bv);
          outw[(size_t)m * D_MODEL + ecol] = __builtin_bit_cast(unsigned short, v);
        }
      } else {
        const int mbase = m0 + 16 * i + 8 * h;       // 8 consecutive rows
        const int bidx = mbase / SEQ, sbase = mbase % SEQ;
        bhalf8 pack;
#pragma unroll
        for (int r = 0; r < 8; ++r) pack[r] = (__bf16)(acc[i][j][r] + bv);
        *reinterpret_cast<bhalf8*>(outw + ((size_t)bidx * D_MODEL + ecol) * SEQ + sbase) = pack;
      }
    }
}

// Flash attention: block = (batch b, 16-query tile qb). 8 waves, wave w owns
// d-slice [w*128, w*128+128). Online softmax state (m, l) per query row in LDS.
__global__ __launch_bounds__(256)
void flash_attn(const unsigned short* __restrict__ qw, const unsigned short* __restrict__ kw,
                const unsigned short* __restrict__ vtw, float* __restrict__ out) {
  const __bf16* Q  = reinterpret_cast<const __bf16*>(qw);
  const __bf16* K  = reinterpret_cast<const __bf16*>(kw);
  const __bf16* Vt = reinterpret_cast<const __bf16*>(vtw);

  __shared__ float sc[16][32];        // score / prob tile
  __shared__ float rowm[16], rowl[16], rowc[16];

  const int tid = threadIdx.x;
  const int lane = tid & 31, w = tid >> 5;
  const int l15 = lane & 15, h = lane >> 4;
  const int qb = blockIdx.x, b = blockIdx.y;
  const int q0 = qb * 16;
  const int dsl = w * 128;
  const float scale = 0.03125f;       // 1/sqrt(1024)

  const __bf16* Qb = Q + (size_t)b * SEQ * D_MODEL;
  const __bf16* Kb = K + (size_t)b * SEQ * D_MODEL;
  const __bf16* Vb = Vt + (size_t)b * D_MODEL * SEQ;

  if (tid < 16) { rowm[tid] = -3.0e38f; rowl[tid] = 0.0f; }

  // Preload Q fragments for this wave's 128-wide d-slice (reused every key tile).
  bhalf16 qf[4];
#pragma unroll
  for (int kk = 0; kk < 4; ++kk)
    qf[kk] = load_afrag(Qb + (size_t)(q0 + l15) * D_MODEL + dsl + 32 * kk, h);

  v8f o[8];
#pragma unroll
  for (int t = 0; t < 8; ++t) o[t] = vzero8();

  const int nkt = qb / 2 + 1;         // causal: key tiles with k0 <= q0+15
  for (int kt = 0; kt < nkt; ++kt) {
    const int k0 = kt * 32;

    // Prefetch next key tile's K rows and V columns (global_prefetch_b8);
    // overlaps next tile's L2/HBM fetch with this tile's softmax + PV.
    if (kt + 1 < nkt) {
      const int kn = k0 + 32;
      __builtin_prefetch(Kb + (size_t)(kn + lane) * D_MODEL + dsl, 0, 1);
      __builtin_prefetch(Vb + (size_t)(dsl + lane * 4) * SEQ + kn, 0, 1);
    }

    __syncthreads();
    reinterpret_cast<float*>(sc)[tid]       = 0.0f;   // zero 512 floats
    reinterpret_cast<float*>(sc)[tid + 256] = 0.0f;
    __syncthreads();

    // Partial scores over this wave's d-slice: S[16q x 32k]
    v8f s0 = vzero8(), s1 = vzero8();
#pragma unroll
    for (int kk = 0; kk < 4; ++kk) {
      bhalf16 kb0 = load_bfrag(Kb + (size_t)(k0 + l15) * D_MODEL + dsl + 32 * kk, h);
      bhalf16 kb1 = load_bfrag(Kb + (size_t)(k0 + 16 + l15) * D_MODEL + dsl + 32 * kk, h);
      s0 = __builtin_amdgcn_wmma_f32_16x16x32_bf16(false, qf[kk], false, kb0, (short)0, s0, false, false);
      s1 = __builtin_amdgcn_wmma_f32_16x16x32_bf16(false, qf[kk], false, kb1, (short)0, s1, false, false);
    }
#pragma unroll
    for (int r = 0; r < 8; ++r) {
      atomicAdd(&sc[r + 8 * h][l15], s0[r]);          // ds_add_f32 cross-wave reduce
      atomicAdd(&sc[r + 8 * h][16 + l15], s1[r]);
    }
    __syncthreads();

    // Online softmax update (one thread per query row).
    if (tid < 16) {
      const int qi = q0 + tid;
      const float mold = rowm[tid];
      float mmax = mold;
      float sv[32];
#pragma unroll
      for (int c = 0; c < 32; ++c) {
        float s = sc[tid][c] * scale;
        if (k0 + c > qi) s = -3.0e38f;                // causal mask
        sv[c] = s;
        mmax = fmaxf(mmax, s);
      }
      const float corr = __expf(mold - mmax);
      float sum = 0.0f;
#pragma unroll
      for (int c = 0; c < 32; ++c) {
        float p = __expf(sv[c] - mmax);
        sc[tid][c] = p;
        sum += p;
      }
      rowc[tid] = corr;
      rowm[tid] = mmax;
      rowl[tid] = rowl[tid] * corr + sum;
    }
    __syncthreads();

    // P fragment (A-role) from LDS, rescale O, then O += P @ V.
    bhalf16 pf;
#pragma unroll
    for (int j = 0; j < 16; ++j) {
      const int c = 16 * (j >> 3) + 8 * h + (j & 7);
      pf[j] = (__bf16)sc[l15][c];
    }
    float cr[8];
#pragma unroll
    for (int r = 0; r < 8; ++r) cr[r] = rowc[r + 8 * h];
#pragma unroll
    for (int t = 0; t < 8; ++t) {
#pragma unroll
      for (int r = 0; r < 8; ++r) o[t][r] *= cr[r];
      const int dt = dsl + 16 * t;
      bhalf16 vf = load_bfrag(Vb + (size_t)(dt + l15) * SEQ + k0, h);
      o[t] = __builtin_amdgcn_wmma_f32_16x16x32_bf16(false, pf, false, vf, (short)0, o[t], false, false);
    }
  }

  // Final 1/l normalization and fp32 store.
  float li[8];
#pragma unroll
  for (int r = 0; r < 8; ++r) li[r] = 1.0f / rowl[r + 8 * h];
#pragma unroll
  for (int t = 0; t < 8; ++t) {
    const int dt = dsl + 16 * t;
#pragma unroll
    for (int r = 0; r < 8; ++r)
      out[((size_t)b * SEQ + q0 + r + 8 * h) * D_MODEL + dt + l15] = o[t][r] * li[r];
  }
}

extern "C" void kernel_launch(void* const* d_in, const int* in_sizes, int n_in,
                              void* d_out, int out_size, void* d_ws, size_t ws_size,
                              hipStream_t stream) {
  (void)in_sizes; (void)n_in; (void)out_size; (void)ws_size;
  const float* x  = (const float*)d_in[0];
  const float* Wq = (const float*)d_in[1];
  const float* bq = (const float*)d_in[2];
  const float* Wk = (const float*)d_in[3];
  const float* bk = (const float*)d_in[4];
  const float* Wv = (const float*)d_in[5];
  const float* bv = (const float*)d_in[6];

  const size_t NX = (size_t)BATCH * SEQ * D_MODEL;   // 8,388,608
  const size_t NW = (size_t)D_MODEL * D_MODEL;       // 1,048,576

  unsigned short* xb  = (unsigned short*)d_ws;
  unsigned short* wqb = xb + NX;
  unsigned short* wkb = wqb + NW;
  unsigned short* wvb = wkb + NW;
  unsigned short* qbf = wvb + NW;
  unsigned short* kbf = qbf + NX;
  unsigned short* vtb = kbf + NX;                    // total ~70 MB of ws

  cvt_f32_to_bf16<<<(int)((NX + 255) / 256), 256, 0, stream>>>(x, xb, (int)NX);
  cvt_f32_to_bf16<<<(int)((NW + 255) / 256), 256, 0, stream>>>(Wq, wqb, (int)NW);
  cvt_f32_to_bf16<<<(int)((NW + 255) / 256), 256, 0, stream>>>(Wk, wkb, (int)NW);
  cvt_f32_to_bf16<<<(int)((NW + 255) / 256), 256, 0, stream>>>(Wv, wvb, (int)NW);

  dim3 ggrid(BATCH * SEQ / 128, D_MODEL / 128);
  qkv_gemm<<<ggrid, 256, 0, stream>>>(xb, wqb, bq, qbf, 0);
  qkv_gemm<<<ggrid, 256, 0, stream>>>(xb, wkb, bk, kbf, 0);
  qkv_gemm<<<ggrid, 256, 0, stream>>>(xb, wvb, bv, vtb, 1);   // V stored transposed [B][D][S]

  flash_attn<<<dim3(SEQ / 16, BATCH), 256, 0, stream>>>(qbf, kbf, vtb, (float*)d_out);
}
